// LSTM_54528904790590
// MI455X (gfx1250) — compile-verified, matrix operand order checked by reading, not run on hardware
//
#include <hip/hip_runtime.h>

// ---------------- problem dims ----------------
#define B_ 64
#define S_ 512
#define I_ 1024
#define H_ 1024
#define NBLK 32     // persistent grid: 32 blocks x 8 waves = 256 waves
#define TPB 256     // == exactly one 16x16 (m,n) patch per wave

// ---------------- workspace layout (bytes) ----------------
#define OFF_WT    0u                         // bf16 W^T [4096][2048]   (16 MB)
#define OFF_BIAS  16777216u                  // f32 bias  [4096]        (16 KB, padded)
#define OFF_X     16842752u                  // bf16 x    [B,S,I]       (64 MB)
#define OFF_H0    83951616u                  // bf16 h ping             (128 KB)
#define OFF_H1    84082688u                  // bf16 h pong             (128 KB)
#define OFF_SYNC  84213760u                  // 2 x u32 barrier state

typedef __attribute__((ext_vector_type(16))) __bf16 v16bf;
typedef __attribute__((ext_vector_type(8)))  float  v8f;

union Frag {
    v16bf v;
    uint4 u[2];
};

__device__ __forceinline__ unsigned short f32_to_bf16(float f) {
    unsigned u = __float_as_uint(f);
    unsigned r = u + 0x7FFFu + ((u >> 16) & 1u);   // round-to-nearest-even
    return (unsigned short)(r >> 16);
}

__device__ __forceinline__ float sigmoid_f(float z) {
    return 1.0f / (1.0f + __expf(-z));
}

// ---------------- prep: pack W^T (bf16) + fused bias ----------------
__global__ void prep_weights(const float* __restrict__ wii, const float* __restrict__ wif,
                             const float* __restrict__ wig, const float* __restrict__ wio,
                             const float* __restrict__ whi, const float* __restrict__ whf,
                             const float* __restrict__ whg, const float* __restrict__ who,
                             const float* __restrict__ bii, const float* __restrict__ bif,
                             const float* __restrict__ big, const float* __restrict__ bio,
                             const float* __restrict__ bhi, const float* __restrict__ bhf,
                             const float* __restrict__ bhg, const float* __restrict__ bho,
                             unsigned short* __restrict__ Wt, float* __restrict__ bias) {
    size_t gid = (size_t)blockIdx.x * blockDim.x + threadIdx.x;
    if (gid >= (size_t)4096 * 2048) return;
    int n    = (int)(gid >> 11);   // 0..4095  (gate-major output column)
    int k    = (int)(gid & 2047);  // 0..2047  (I rows then H rows)
    int gate = n >> 10;            // 0:i 1:f 2:g 3:o
    int col  = n & 1023;
    const float* Wi[4] = {wii, wif, wig, wio};
    const float* Wh[4] = {whi, whf, whg, who};
    float v = (k < 1024) ? Wi[gate][(size_t)k * H_ + col]
                         : Wh[gate][(size_t)(k - 1024) * H_ + col];
    Wt[gid] = f32_to_bf16(v);
    if (k == 0) {
        const float* bi[4] = {bii, bif, big, bio};
        const float* bh[4] = {bhi, bhf, bhg, bho};
        bias[n] = bi[gate][col] + bh[gate][col];
    }
}

// ---------------- prep: x fp32 -> bf16 ----------------
__global__ void prep_x(const float* __restrict__ x, unsigned short* __restrict__ xb) {
    size_t gid = (size_t)blockIdx.x * blockDim.x + threadIdx.x;
    if (gid < (size_t)B_ * S_ * I_) xb[gid] = f32_to_bf16(x[gid]);
}

// ---------------- prep: zero h ping-pong + barrier state ----------------
__global__ void init_state(unsigned short* __restrict__ h0, unsigned short* __restrict__ h1,
                           unsigned* __restrict__ syncv) {
    size_t gid = (size_t)blockIdx.x * blockDim.x + threadIdx.x;
    if (gid < (size_t)B_ * H_) { h0[gid] = 0; h1[gid] = 0; }
    if (gid < 2) syncv[gid] = 0u;
}

// ---------------- grid-wide barrier (persistent kernel) ----------------
__device__ __forceinline__ void grid_sync(unsigned* cnt, unsigned* gen) {
    __threadfence();            // make this thread's h stores device-visible
    __syncthreads();
    if (threadIdx.x == 0) {
        unsigned g = __hip_atomic_load(gen, __ATOMIC_ACQUIRE, __HIP_MEMORY_SCOPE_AGENT);
        unsigned a = __hip_atomic_fetch_add(cnt, 1u, __ATOMIC_ACQ_REL, __HIP_MEMORY_SCOPE_AGENT);
        if (a == NBLK - 1u) {
            __hip_atomic_store(cnt, 0u, __ATOMIC_RELAXED, __HIP_MEMORY_SCOPE_AGENT);
            __hip_atomic_fetch_add(gen, 1u, __ATOMIC_ACQ_REL, __HIP_MEMORY_SCOPE_AGENT);
        } else {
            while (__hip_atomic_load(gen, __ATOMIC_ACQUIRE, __HIP_MEMORY_SCOPE_AGENT) == g) {
                __builtin_amdgcn_s_sleep(8);
            }
        }
    }
    __syncthreads();
    __threadfence();
}

// ---------------- persistent fused LSTM scan ----------------
// Each wave owns output patch (m0:m0+16, n0:n0+16) for ALL 512 timesteps.
// Per step: z[4 gates] = [x_t | h] @ W^T via bf16 WMMA, fp32 accum; gates +
// cell update fully in registers; only h round-trips memory (bf16 ping-pong).
__global__ void __launch_bounds__(TPB)
lstm_persistent(const unsigned short* __restrict__ xb,   // bf16 [B,S,I]
                const unsigned short* __restrict__ Wt,   // bf16 [4096][2048]
                const float* __restrict__ bias,          // f32  [4096]
                unsigned short* __restrict__ h0,         // bf16 [B,H]
                unsigned short* __restrict__ h1,         // bf16 [B,H]
                float* __restrict__ out,                 // f32  [2*B*H] = h_f, c_f
                unsigned* __restrict__ syncv) {
    const int lane = threadIdx.x & 31;
    const int wave = blockIdx.x * (TPB >> 5) + (threadIdx.x >> 5);  // 0..255
    const int m0   = (wave & 3) << 4;    // batch tile   (64/16 = 4)
    const int n0   = (wave >> 2) << 4;   // column tile  (1024/16 = 64)
    const int half = lane >> 4;          // K-half select (ISA 16-bit A layout)
    const int lm   = lane & 15;
    const int mrow = m0 + lm;            // row this lane loads for A
    const int ncol = n0 + lm;            // output column (within each gate)
    const int koff = half * 8;           // lanes 16-31 read K+8 .. and K+24 ..

    const float bi = bias[ncol];
    const float bf = bias[1024 + ncol];
    const float bg = bias[2048 + ncol];
    const float bo = bias[3072 + ncol];

    // B-fragment row pointers: W^T row = gate*1024 + (n0 + lm), contiguous in K
    const unsigned short* w0 = Wt + (size_t)(0 * 1024 + n0 + lm) * 2048 + koff;
    const unsigned short* w1 = Wt + (size_t)(1 * 1024 + n0 + lm) * 2048 + koff;
    const unsigned short* w2 = Wt + (size_t)(2 * 1024 + n0 + lm) * 2048 + koff;
    const unsigned short* w3 = Wt + (size_t)(3 * 1024 + n0 + lm) * 2048 + koff;

    const unsigned short* xrow_base = xb + (size_t)mrow * S_ * I_ + koff;

    float creg[8];
#pragma unroll
    for (int e = 0; e < 8; ++e) creg[e] = 0.0f;

    unsigned* cnt = syncv;
    unsigned* gen = syncv + 1;

    for (int t = 0; t < S_; ++t) {
        const unsigned short* hread  = (t & 1) ? h1 : h0;
        unsigned short*       hwrite = (t & 1) ? h0 : h1;
        const unsigned short* xrow = xrow_base + (size_t)t * I_;
        const unsigned short* hrow = hread + (size_t)mrow * H_ + koff;

        v8f acc0 = {}, acc1 = {}, acc2 = {}, acc3 = {};

        // ---- K part 1: input projection (A rows from x_t) ----
#pragma unroll 4
        for (int kk = 0; kk < 1024; kk += 32) {
            Frag a, b0, b1, b2, b3;
            a.u[0]  = *(const uint4*)(xrow + kk);
            a.u[1]  = *(const uint4*)(xrow + kk + 16);
            b0.u[0] = *(const uint4*)(w0 + kk);  b0.u[1] = *(const uint4*)(w0 + kk + 16);
            b1.u[0] = *(const uint4*)(w1 + kk);  b1.u[1] = *(const uint4*)(w1 + kk + 16);
            b2.u[0] = *(const uint4*)(w2 + kk);  b2.u[1] = *(const uint4*)(w2 + kk + 16);
            b3.u[0] = *(const uint4*)(w3 + kk);  b3.u[1] = *(const uint4*)(w3 + kk + 16);
            acc0 = __builtin_amdgcn_wmma_f32_16x16x32_bf16(false, a.v, false, b0.v, (short)0, acc0, false, false);
            acc1 = __builtin_amdgcn_wmma_f32_16x16x32_bf16(false, a.v, false, b1.v, (short)0, acc1, false, false);
            acc2 = __builtin_amdgcn_wmma_f32_16x16x32_bf16(false, a.v, false, b2.v, (short)0, acc2, false, false);
            acc3 = __builtin_amdgcn_wmma_f32_16x16x32_bf16(false, a.v, false, b3.v, (short)0, acc3, false, false);
        }
        // ---- K part 2: recurrent projection (A rows from h_{t-1}) ----
#pragma unroll 4
        for (int kk = 0; kk < 1024; kk += 32) {
            Frag a, b0, b1, b2, b3;
            a.u[0]  = *(const uint4*)(hrow + kk);
            a.u[1]  = *(const uint4*)(hrow + kk + 16);
            b0.u[0] = *(const uint4*)(w0 + 1024 + kk);  b0.u[1] = *(const uint4*)(w0 + 1024 + kk + 16);
            b1.u[0] = *(const uint4*)(w1 + 1024 + kk);  b1.u[1] = *(const uint4*)(w1 + 1024 + kk + 16);
            b2.u[0] = *(const uint4*)(w2 + 1024 + kk);  b2.u[1] = *(const uint4*)(w2 + 1024 + kk + 16);
            b3.u[0] = *(const uint4*)(w3 + 1024 + kk);  b3.u[1] = *(const uint4*)(w3 + 1024 + kk + 16);
            acc0 = __builtin_amdgcn_wmma_f32_16x16x32_bf16(false, a.v, false, b0.v, (short)0, acc0, false, false);
            acc1 = __builtin_amdgcn_wmma_f32_16x16x32_bf16(false, a.v, false, b1.v, (short)0, acc1, false, false);
            acc2 = __builtin_amdgcn_wmma_f32_16x16x32_bf16(false, a.v, false, b2.v, (short)0, acc2, false, false);
            acc3 = __builtin_amdgcn_wmma_f32_16x16x32_bf16(false, a.v, false, b3.v, (short)0, acc3, false, false);
        }

        // ---- gates + cell update, fully in registers ----
        // C/D layout: VGPR e, lanes 0-15 -> M=e, lanes 16-31 -> M=8+e; N=lane%16
#pragma unroll
        for (int e = 0; e < 8; ++e) {
            int m = m0 + half * 8 + e;
            size_t idx = (size_t)m * H_ + ncol;
            float it = sigmoid_f(acc0[e] + bi);
            float ft = sigmoid_f(acc1[e] + bf);
            float gt = tanhf(acc2[e] + bg);
            float ot = sigmoid_f(acc3[e] + bo);
            float cn = ft * creg[e] + it * gt;
            float hn = ot * tanhf(cn);
            creg[e] = cn;
            hwrite[idx] = f32_to_bf16(hn);
            if (t == S_ - 1) {
                out[idx] = hn;                       // h_f
                out[(size_t)B_ * H_ + idx] = cn;     // c_f
            }
        }

        if (t != S_ - 1) grid_sync(cnt, gen);
    }
}

// ---------------- host-side launch ----------------
extern "C" void kernel_launch(void* const* d_in, const int* in_sizes, int n_in,
                              void* d_out, int out_size, void* d_ws, size_t ws_size,
                              hipStream_t stream) {
    const float* x   = (const float*)d_in[0];
    const float* wii = (const float*)d_in[1];
    const float* wif = (const float*)d_in[2];
    const float* wig = (const float*)d_in[3];
    const float* wio = (const float*)d_in[4];
    const float* whi = (const float*)d_in[5];
    const float* whf = (const float*)d_in[6];
    const float* whg = (const float*)d_in[7];
    const float* who = (const float*)d_in[8];
    const float* bii = (const float*)d_in[9];
    const float* bif = (const float*)d_in[10];
    const float* big = (const float*)d_in[11];
    const float* bio = (const float*)d_in[12];
    const float* bhi = (const float*)d_in[13];
    const float* bhf = (const float*)d_in[14];
    const float* bhg = (const float*)d_in[15];
    const float* bho = (const float*)d_in[16];

    char* ws = (char*)d_ws;
    unsigned short* Wt   = (unsigned short*)(ws + OFF_WT);
    float*          bias = (float*)         (ws + OFF_BIAS);
    unsigned short* xb   = (unsigned short*)(ws + OFF_X);
    unsigned short* h0   = (unsigned short*)(ws + OFF_H0);
    unsigned short* h1   = (unsigned short*)(ws + OFF_H1);
    unsigned*       sv   = (unsigned*)      (ws + OFF_SYNC);

    {   // W^T + bias pack
        size_t nelem = (size_t)4096 * 2048;
        int blocks = (int)((nelem + 255) / 256);
        prep_weights<<<blocks, 256, 0, stream>>>(wii, wif, wig, wio, whi, whf, whg, who,
                                                 bii, bif, big, bio, bhi, bhf, bhg, bho,
                                                 Wt, bias);
    }
    {   // x -> bf16
        size_t nelem = (size_t)B_ * S_ * I_;
        int blocks = (int)((nelem + 255) / 256);
        prep_x<<<blocks, 256, 0, stream>>>(x, xb);
    }
    {   // zero h ping-pong + barrier counters (re-done every call: deterministic)
        size_t nelem = (size_t)B_ * H_;
        int blocks = (int)((nelem + 255) / 256);
        init_state<<<blocks, 256, 0, stream>>>(h0, h1, sv);
    }
    lstm_persistent<<<NBLK, TPB, 0, stream>>>(xb, Wt, bias, h0, h1, (float*)d_out, sv);
}
